// Discriminator_RNN_Utt_Jit_40037685133440
// MI455X (gfx1250) — matrix operand order repaired and checked by reading, not compiled, not run on hardware
//
#include <hip/hip_runtime.h>
#include <hip/hip_bf16.h>

// ---------------------------------------------------------------------------
// Discriminator_RNN_Utt: 2-layer GRU (B=1024, T=63, F=257, H=512) + MLP heads
// bf16 WMMA (v_wmma_f32_16x16x32_bf16) GEMMs, f32 accumulate.
// Per-timestep xg/hg GEMMs + fused gate kernel.
// GEMM: 32x64 tile per wave (8 WMMA / K-chunk); K is a template constant
// (288 or 512) so the chunk loop fully unrolls: fragment loads become
// immediate-offset b128 loads off loop-invariant bases, no register
// rotation movs, no per-iteration address VALU.
// ---------------------------------------------------------------------------

#define B_DIM 1024
#define T_DIM 63
#define F_DIM 257
#define KP0   288      // F padded to multiple of 32
#define H_DIM 512
#define H3    1536     // 3*H

typedef __bf16 bf16_t;
typedef __attribute__((ext_vector_type(16))) __bf16 v16bf;
typedef __attribute__((ext_vector_type(8)))  __bf16 bf16x8;
typedef __attribute__((ext_vector_type(8)))  float  v8f;

__device__ inline v16bf join16(bf16x8 lo, bf16x8 hi) {
    union { v16bf v; bf16x8 h[2]; } u;
    u.h[0] = lo; u.h[1] = hi;
    return u.v;
}

// ---------------------------------------------------------------------------
// C[M=1024, N] (f32) = A[M x K](bf16, row stride strideA) * W[N x K](bf16)^T
//                      + bias[N]
// Block: 256 threads = 8 waves, arranged 4 (M) x 2 (N) -> 128x128 block tile.
// Wave: 32x64 strip = 2x4 WMMA tiles. K fully unrolled in 32-chunks.
// A-fragment: lane l<16 -> row M0+l, K = k..k+15
//             lane l>=16 -> row M0+l-16, K = k+16..k+31
// B-fragment: lane n<16 -> col N0+n, K = k..k+15 (row n of W, contiguous)
//             lane n>=16 -> col N0+n-16, K = k+16..k+31
// ---------------------------------------------------------------------------
template <int K>
__global__ void __launch_bounds__(256)
wmma_gemm_bias(const bf16_t* __restrict__ A, int strideA,
               const bf16_t* __restrict__ W,
               const float* __restrict__ bias,
               float* __restrict__ C, int N)
{
    const int lane   = threadIdx.x & 31;
    const int wave   = threadIdx.x >> 5;
    const int la     = lane & 15;
    const int hiHalf = lane >> 4;                       // 0: lanes 0-15, 1: 16-31
    const int m0     = blockIdx.y * 128 + (wave & 3) * 32;
    const int n0     = blockIdx.x * 128 + (wave >> 2) * 64;

    v8f acc[2][4];
#pragma unroll
    for (int mi = 0; mi < 2; ++mi)
#pragma unroll
        for (int j = 0; j < 4; ++j) acc[mi][j] = v8f{};

    // Loop-invariant fragment base pointers; all K offsets are immediates.
    const bf16_t* aPtr0 = A + (size_t)(m0 + la) * strideA + hiHalf * 16;
    const bf16_t* aPtr1 = aPtr0 + (size_t)16 * strideA;
    const bf16_t* bPtr0 = W + (size_t)(n0 + 0  + la) * K + hiHalf * 16;
    const bf16_t* bPtr1 = W + (size_t)(n0 + 16 + la) * K + hiHalf * 16;
    const bf16_t* bPtr2 = W + (size_t)(n0 + 32 + la) * K + hiHalf * 16;
    const bf16_t* bPtr3 = W + (size_t)(n0 + 48 + la) * K + hiHalf * 16;

    constexpr int CHUNKS = K / 32;
#pragma unroll
    for (int c = 0; c < CHUNKS; ++c) {
        const int k = c * 32;
        v16bf a0 = join16(*(const bf16x8*)(aPtr0 + k), *(const bf16x8*)(aPtr0 + k + 8));
        v16bf a1 = join16(*(const bf16x8*)(aPtr1 + k), *(const bf16x8*)(aPtr1 + k + 8));
        v16bf b0 = join16(*(const bf16x8*)(bPtr0 + k), *(const bf16x8*)(bPtr0 + k + 8));
        v16bf b1 = join16(*(const bf16x8*)(bPtr1 + k), *(const bf16x8*)(bPtr1 + k + 8));
        v16bf b2 = join16(*(const bf16x8*)(bPtr2 + k), *(const bf16x8*)(bPtr2 + k + 8));
        v16bf b3 = join16(*(const bf16x8*)(bPtr3 + k), *(const bf16x8*)(bPtr3 + k + 8));

        acc[0][0] = __builtin_amdgcn_wmma_f32_16x16x32_bf16(false, a0, false, b0, (short)0, acc[0][0], false, false);
        acc[0][1] = __builtin_amdgcn_wmma_f32_16x16x32_bf16(false, a0, false, b1, (short)0, acc[0][1], false, false);
        acc[0][2] = __builtin_amdgcn_wmma_f32_16x16x32_bf16(false, a0, false, b2, (short)0, acc[0][2], false, false);
        acc[0][3] = __builtin_amdgcn_wmma_f32_16x16x32_bf16(false, a0, false, b3, (short)0, acc[0][3], false, false);
        acc[1][0] = __builtin_amdgcn_wmma_f32_16x16x32_bf16(false, a1, false, b0, (short)0, acc[1][0], false, false);
        acc[1][1] = __builtin_amdgcn_wmma_f32_16x16x32_bf16(false, a1, false, b1, (short)0, acc[1][1], false, false);
        acc[1][2] = __builtin_amdgcn_wmma_f32_16x16x32_bf16(false, a1, false, b2, (short)0, acc[1][2], false, false);
        acc[1][3] = __builtin_amdgcn_wmma_f32_16x16x32_bf16(false, a1, false, b3, (short)0, acc[1][3], false, false);
    }

    // ---- store with bias ----
#pragma unroll
    for (int j = 0; j < 4; ++j) {
        const int n  = n0 + j * 16 + la;
        const float bv = bias[n];
#pragma unroll
        for (int mi = 0; mi < 2; ++mi) {
#pragma unroll
            for (int r = 0; r < 8; ++r) {
                const int row = m0 + mi * 16 + r + hiHalf * 8;
                C[(size_t)row * N + n] = acc[mi][j][r] + bv;
            }
        }
    }
}

// ---------------------------------------------------------------------------
// f32 -> bf16 convert with zero K-padding (rows x Kin -> rows x Kp)
// ---------------------------------------------------------------------------
__global__ void cvt_pad_bf16(const float* __restrict__ src, bf16_t* __restrict__ dst,
                             int rows, int Kin, int Kp)
{
    int idx = blockIdx.x * blockDim.x + threadIdx.x;
    if (idx >= rows * Kp) return;
    int r = idx / Kp, k = idx - r * Kp;
    float v = (k < Kin) ? src[(size_t)r * Kin + k] : 0.f;
    dst[idx] = (bf16_t)v;
}

__global__ void zero_h(float* __restrict__ hf, bf16_t* __restrict__ hb)
{
    int idx = blockIdx.x * blockDim.x + threadIdx.x;
    if (idx < B_DIM * H_DIM) { hf[idx] = 0.f; hb[idx] = (bf16_t)0.f; }
}

// ---------------------------------------------------------------------------
// Fused GRU gates: r,z,n -> h_new = (1-z)*n + z*h. Updates f32 + bf16 state
// in place and streams bf16 layer output into seq[:, t, :].
// ---------------------------------------------------------------------------
__global__ void gru_gate_kernel(const float* __restrict__ xg,
                                const float* __restrict__ hg,
                                float* __restrict__ hstate_f32,
                                bf16_t* __restrict__ hstate_bf,
                                bf16_t* __restrict__ seq_out,
                                int t)
{
    int idx = blockIdx.x * blockDim.x + threadIdx.x;   // [0, B*H)
    if (idx >= B_DIM * H_DIM) return;
    int b = idx >> 9;          // / 512
    int j = idx & (H_DIM - 1);
    const float* xr = xg + (size_t)b * H3;
    const float* hr = hg + (size_t)b * H3;
    float i_r = xr[j], i_z = xr[H_DIM + j], i_n = xr[2 * H_DIM + j];
    float h_r = hr[j], h_z = hr[H_DIM + j], h_n = hr[2 * H_DIM + j];
    float r = 1.f / (1.f + expf(-(i_r + h_r)));
    float z = 1.f / (1.f + expf(-(i_z + h_z)));
    float n = tanhf(i_n + r * h_n);
    float h = hstate_f32[idx];
    float hn = n - n * z + z * h;
    hstate_f32[idx] = hn;
    bf16_t hb = (bf16_t)hn;
    hstate_bf[idx] = hb;
    seq_out[((size_t)b * T_DIM + t) * H_DIM + j] = hb;
}

// ---------------------------------------------------------------------------
// x3[b,t] = dot(h2[b,t,:], dnn_w) + dnn_b  -- one wave per (b,t) row,
// contiguous 16B bf16x8 loads per lane.
// ---------------------------------------------------------------------------
__global__ void dnn_head(const bf16_t* __restrict__ h2, const float* __restrict__ dw,
                         const float* __restrict__ db, float* __restrict__ x3)
{
    int gid  = blockIdx.x * blockDim.x + threadIdx.x;
    int row  = gid >> 5;
    int lane = gid & 31;
    if (row >= B_DIM * T_DIM) return;
    const bf16_t* hr = h2 + (size_t)row * H_DIM;
    float s = 0.f;
#pragma unroll
    for (int base = 0; base < H_DIM; base += 32 * 8) {
        const int o = base + lane * 8;
        bf16x8 v = *(const bf16x8*)(hr + o);
#pragma unroll
        for (int e = 0; e < 8; ++e) s += (float)v[e] * dw[o + e];
    }
#pragma unroll
    for (int off = 16; off > 0; off >>= 1) s += __shfl_down(s, off, 32);
    if (lane == 0) x3[row] = s + db[0];
}

// ---------------------------------------------------------------------------
// Tiny MLP heads: [B,63] -> 32 -> 16 -> 1 (all linear, no activation)
// ---------------------------------------------------------------------------
__global__ void mlp_head(const float* __restrict__ x3,
                         const float* __restrict__ w1, const float* __restrict__ b1,
                         const float* __restrict__ w2, const float* __restrict__ b2,
                         const float* __restrict__ w3, const float* __restrict__ b3,
                         float* __restrict__ out)
{
    int b = blockIdx.x * blockDim.x + threadIdx.x;
    if (b >= B_DIM) return;
    const float* xr = x3 + (size_t)b * T_DIM;
    float x4[32];
#pragma unroll 4
    for (int o = 0; o < 32; ++o) {
        float s = b1[o];
        const float* wr = w1 + o * T_DIM;
        for (int t = 0; t < T_DIM; ++t) s += xr[t] * wr[t];
        x4[o] = s;
    }
    float x5[16];
#pragma unroll
    for (int o = 0; o < 16; ++o) {
        float s = b2[o];
#pragma unroll
        for (int i = 0; i < 32; ++i) s += x4[i] * w2[o * 32 + i];
        x5[o] = s;
    }
    float s = b3[0];
#pragma unroll
    for (int i = 0; i < 16; ++i) s += x5[i] * w3[i];
    out[b] = s;
}

// ---------------------------------------------------------------------------
extern "C" void kernel_launch(void* const* d_in, const int* in_sizes, int n_in,
                              void* d_out, int out_size, void* d_ws, size_t ws_size,
                              hipStream_t stream)
{
    const float* x     = (const float*)d_in[0];
    const float* w_ih0 = (const float*)d_in[1];
    const float* w_hh0 = (const float*)d_in[2];
    const float* b_ih0 = (const float*)d_in[3];
    const float* b_hh0 = (const float*)d_in[4];
    const float* w_ih1 = (const float*)d_in[5];
    const float* w_hh1 = (const float*)d_in[6];
    const float* b_ih1 = (const float*)d_in[7];
    const float* b_hh1 = (const float*)d_in[8];
    const float* dnn_w = (const float*)d_in[9];
    const float* dnn_b = (const float*)d_in[10];
    const float* w1    = (const float*)d_in[11];
    const float* b1    = (const float*)d_in[12];
    const float* w2    = (const float*)d_in[13];
    const float* b2    = (const float*)d_in[14];
    const float* w3    = (const float*)d_in[15];
    const float* b3    = (const float*)d_in[16];
    (void)in_sizes; (void)n_in; (void)out_size; (void)ws_size;

    // Workspace carve-out (~120 MB total)
    char* p = (char*)d_ws;
    auto carve = [&](size_t bytes) {
        char* r = p;
        p += (bytes + 255) & ~(size_t)255;
        return (void*)r;
    };
    bf16_t* x_bf    = (bf16_t*)carve((size_t)B_DIM * T_DIM * KP0 * 2);   // padded x
    bf16_t* seq_bf  = (bf16_t*)carve((size_t)B_DIM * T_DIM * H_DIM * 2); // h1 then h2
    bf16_t* wih0_bf = (bf16_t*)carve((size_t)H3 * KP0 * 2);
    bf16_t* whh0_bf = (bf16_t*)carve((size_t)H3 * H_DIM * 2);
    bf16_t* wih1_bf = (bf16_t*)carve((size_t)H3 * H_DIM * 2);
    bf16_t* whh1_bf = (bf16_t*)carve((size_t)H3 * H_DIM * 2);
    float*  xg      = (float*)carve((size_t)B_DIM * H3 * 4);
    float*  hg      = (float*)carve((size_t)B_DIM * H3 * 4);
    float*  hf      = (float*)carve((size_t)B_DIM * H_DIM * 4);
    bf16_t* hb      = (bf16_t*)carve((size_t)B_DIM * H_DIM * 2);
    float*  x3      = (float*)carve((size_t)B_DIM * T_DIM * 4);

    // One-time bf16 conversions (weights stay resident in L2 afterwards)
    {
        int n = B_DIM * T_DIM * KP0;
        cvt_pad_bf16<<<(n + 255) / 256, 256, 0, stream>>>(x, x_bf, B_DIM * T_DIM, F_DIM, KP0);
        n = H3 * KP0;
        cvt_pad_bf16<<<(n + 255) / 256, 256, 0, stream>>>(w_ih0, wih0_bf, H3, F_DIM, KP0);
        n = H3 * H_DIM;
        cvt_pad_bf16<<<(n + 255) / 256, 256, 0, stream>>>(w_hh0, whh0_bf, H3, H_DIM, H_DIM);
        cvt_pad_bf16<<<(n + 255) / 256, 256, 0, stream>>>(w_ih1, wih1_bf, H3, H_DIM, H_DIM);
        cvt_pad_bf16<<<(n + 255) / 256, 256, 0, stream>>>(w_hh1, whh1_bf, H3, H_DIM, H_DIM);
    }

    const dim3 gGemm(H3 / 128, B_DIM / 128);  // (12, 8) blocks, 256 thr each
    const int  nGate = B_DIM * H_DIM;
    const int  gGate = (nGate + 255) / 256;

    // ---- GRU layer 0: x[B,T,Fpad] -> seq_bf (h1) ----
    zero_h<<<gGate, 256, 0, stream>>>(hf, hb);
    for (int t = 0; t < T_DIM; ++t) {
        wmma_gemm_bias<KP0><<<gGemm, 256, 0, stream>>>(
            x_bf + (size_t)t * KP0, T_DIM * KP0, wih0_bf, b_ih0, xg, H3);
        wmma_gemm_bias<H_DIM><<<gGemm, 256, 0, stream>>>(
            hb, H_DIM, whh0_bf, b_hh0, hg, H3);
        gru_gate_kernel<<<gGate, 256, 0, stream>>>(xg, hg, hf, hb, seq_bf, t);
    }

    // ---- GRU layer 1: seq_bf (h1) -> seq_bf (h2, overwritten per-t after use) ----
    zero_h<<<gGate, 256, 0, stream>>>(hf, hb);
    for (int t = 0; t < T_DIM; ++t) {
        wmma_gemm_bias<H_DIM><<<gGemm, 256, 0, stream>>>(
            seq_bf + (size_t)t * H_DIM, T_DIM * H_DIM, wih1_bf, b_ih1, xg, H3);
        wmma_gemm_bias<H_DIM><<<gGemm, 256, 0, stream>>>(
            hb, H_DIM, whh1_bf, b_hh1, hg, H3);
        gru_gate_kernel<<<gGate, 256, 0, stream>>>(xg, hg, hf, hb, seq_bf, t);
    }

    // ---- heads ----
    {
        int waves = B_DIM * T_DIM;                        // one wave per (b,t)
        int thr   = waves * 32;
        dnn_head<<<(thr + 255) / 256, 256, 0, stream>>>(seq_bf, dnn_w, dnn_b, x3);
        mlp_head<<<(B_DIM + 255) / 256, 256, 0, stream>>>(x3, w1, b1, w2, b2, w3, b3,
                                                          (float*)d_out);
    }
}
